// GATLayer_38482906972560
// MI455X (gfx1250) — compile-verified, matrix operand order checked by reading, not run on hardware
//
#include <hip/hip_runtime.h>
#include <hip/hip_bf16.h>

#define N_NODES 50000
#define N_EDGES 800000
#define IN_F 128
#define OUT_F 64
#define HEADS 4
#define HF 256            // HEADS * OUT_F
#define NEG_SLOPE 0.2f
#define GAT_EPS 1e-16f

typedef __attribute__((ext_vector_type(16))) _Float16 v16h;
typedef __attribute__((ext_vector_type(8)))  _Float16 v8h;
typedef __attribute__((ext_vector_type(8)))  float    v8f;
typedef __attribute__((ext_vector_type(4)))  float    v4f;

// ---------------------------------------------------------------------------
// Kernel 1: convert x and W to f16 for WMMA consumption
// ---------------------------------------------------------------------------
__global__ void cvt_f16_kernel(const float* __restrict__ x, const float* __restrict__ W,
                               _Float16* __restrict__ Xh, _Float16* __restrict__ Wh) {
  int idx = blockIdx.x * blockDim.x + threadIdx.x;
  int stride = gridDim.x * blockDim.x;
  for (int i = idx; i < N_NODES * IN_F; i += stride) Xh[i] = (_Float16)x[i];
  for (int i = idx; i < HF * IN_F; i += stride)      Wh[i] = (_Float16)W[i];
}

// ---------------------------------------------------------------------------
// Kernel 2: init out = bias (per row), m = -inf encoding, s = 0
// ---------------------------------------------------------------------------
__global__ void init_kernel(float* __restrict__ out, const float* __restrict__ bias,
                            int* __restrict__ m, float* __restrict__ ssum) {
  int idx = blockIdx.x * blockDim.x + threadIdx.x;
  int stride = gridDim.x * blockDim.x;
  for (int i = idx; i < N_NODES * HF; i += stride) out[i] = bias[i & (HF - 1)];
  for (int i = idx; i < N_NODES * HEADS; i += stride) {
    m[i] = (int)0x80000000;   // < encoding of any float, incl -inf
    ssum[i] = 0.0f;
  }
}

// ---------------------------------------------------------------------------
// Kernel 3: h = x @ W^T via v_wmma_f32_16x16x32_f16.
// One wave per 16x16 output tile. waveId = tileM*16 + tileN; K-loop 4 x 32.
// N_NODES (50000) is divisible by 16, so all 3125 M-tiles are fully in-bounds:
// no clamps, no predicated stores -> straight-line epilogue, EXEC all-ones.
// 16-bit A fragment layout (ISA 7.12.2): lane<16 holds row M=lane,
//   elements 0-7 = K 0-7, 8-15 = K 16-23 of the 32-chunk; lane>=16 same rows,
//   K offset by 8.  B loaded identically from W rows (col-of-B == row-of-W).
// ---------------------------------------------------------------------------
__global__ void __launch_bounds__(256)
gemm_wmma_kernel(const _Float16* __restrict__ Xh, const _Float16* __restrict__ Wh,
                 float* __restrict__ h) {
  const int wave  = (blockIdx.x << 3) + (threadIdx.x >> 5);   // 0..49999
  const int lane  = threadIdx.x & 31;
  const int tileM = wave >> 4;
  const int tileN = wave & 15;
  const int mrow0 = tileM << 4;
  const int ncol0 = tileN << 4;

  const int l15   = lane & 15;
  const int khalf = lane >> 4;                                 // 0 or 1

  const int arow = mrow0 + l15;                                // always < 50000
  const int brow = ncol0 + l15;                                // always < 256

  const _Float16* ap = Xh + (size_t)arow * IN_F + khalf * 8;
  const _Float16* bp = Wh + (size_t)brow * IN_F + khalf * 8;

  v8f c = {};
  #pragma unroll
  for (int kb = 0; kb < IN_F; kb += 32) {
    v8h a0 = *(const v8h*)(ap + kb);
    v8h a1 = *(const v8h*)(ap + kb + 16);
    v8h b0 = *(const v8h*)(bp + kb);
    v8h b1 = *(const v8h*)(bp + kb + 16);
    v16h A, B;
    #pragma unroll
    for (int i = 0; i < 8; ++i) { A[i] = a0[i]; A[i + 8] = a1[i];
                                  B[i] = b0[i]; B[i + 8] = b1[i]; }
    c = __builtin_amdgcn_wmma_f32_16x16x32_f16(false, A, false, B, (short)0, c,
                                               false, false);
  }
  // D layout: VGPR r, lane -> (M = mrow0 + r + 8*khalf, N = ncol0 + l15)
  float* hp = h + (size_t)(mrow0 + khalf * 8) * HF + ncol0 + l15;
  #pragma unroll
  for (int r = 0; r < 8; ++r) hp[(size_t)r * HF] = c[r];
}

// ---------------------------------------------------------------------------
// Kernel 4: per-node attention logits.  One wave per node; lane l covers the
// 8 contiguous features [8l, 8l+8) which all belong to head l>>3; reduce over
// each group of 8 lanes with shfl_xor.
// ---------------------------------------------------------------------------
__global__ void __launch_bounds__(256)
logits_kernel(const float* __restrict__ h,
              const float* __restrict__ att_src, const float* __restrict__ att_dst,
              float* __restrict__ a_src, float* __restrict__ a_dst) {
  int node = (blockIdx.x << 3) + (threadIdx.x >> 5);
  if (node >= N_NODES) return;
  int lane  = threadIdx.x & 31;
  int head  = lane >> 3;
  int fbase = (lane & 7) * 8;                       // feature offset inside head
  const float* hp = h + (size_t)node * HF + lane * 8;
  float ps = 0.f, pd = 0.f;
  #pragma unroll
  for (int j = 0; j < 8; ++j) {
    float hv = hp[j];
    ps += hv * att_src[head * OUT_F + fbase + j];
    pd += hv * att_dst[head * OUT_F + fbase + j];
  }
  #pragma unroll
  for (int off = 1; off < 8; off <<= 1) {
    ps += __shfl_xor(ps, off, 32);
    pd += __shfl_xor(pd, off, 32);
  }
  if ((lane & 7) == 0) {
    a_src[node * HEADS + head] = ps;
    a_dst[node * HEADS + head] = pd;
  }
}

// Monotonic int encoding of float for hardware integer atomicMax
__device__ __forceinline__ int enc_f(float f) {
  int i = __float_as_int(f);
  return (i >= 0) ? i : (i ^ 0x7fffffff);
}
__device__ __forceinline__ float dec_f(int i) {
  return __int_as_float((i >= 0) ? i : (i ^ 0x7fffffff));
}

// ---------------------------------------------------------------------------
// Kernel 5: per-edge raw score + LeakyReLU; segment max via int atomicMax
// ---------------------------------------------------------------------------
__global__ void edge_score_kernel(const int* __restrict__ ei,
                                  const float* __restrict__ a_src,
                                  const float* __restrict__ a_dst,
                                  float* __restrict__ ebuf, int* __restrict__ m) {
  int i = blockIdx.x * blockDim.x + threadIdx.x;
  if (i >= N_EDGES) return;
  int s = ei[i], d = ei[N_EDGES + i];
  #pragma unroll
  for (int hh = 0; hh < HEADS; ++hh) {
    float ev = a_src[s * HEADS + hh] + a_dst[d * HEADS + hh];
    ev = (ev >= 0.f) ? ev : NEG_SLOPE * ev;
    ebuf[(size_t)i * HEADS + hh] = ev;
    atomicMax(&m[d * HEADS + hh], enc_f(ev));
  }
}

// ---------------------------------------------------------------------------
// Kernel 6: ex = exp(e - m[dst]); segment sum via f32 hardware atomics
// ---------------------------------------------------------------------------
__global__ void edge_exp_kernel(const int* __restrict__ ei, const int* __restrict__ m,
                                float* __restrict__ ebuf, float* __restrict__ ssum) {
  int i = blockIdx.x * blockDim.x + threadIdx.x;
  if (i >= N_EDGES) return;
  int d = ei[N_EDGES + i];
  #pragma unroll
  for (int hh = 0; hh < HEADS; ++hh) {
    int menc = m[d * HEADS + hh];
    float mv = (menc == (int)0x80000000) ? 0.f : dec_f(menc);
    float ex = __expf(ebuf[(size_t)i * HEADS + hh] - mv);
    ebuf[(size_t)i * HEADS + hh] = ex;
    atomicAdd(&ssum[d * HEADS + hh], ex);
  }
}

// ---------------------------------------------------------------------------
// Kernel 7: weighted scatter-aggregate.  One wave per edge; lane l handles
// features {l, l+32, ...} so each of the 8 bursts is 128 B contiguous.  The
// h/out working set (~102 MB) lives in the 192 MB L2, so the f32 atomic adds
// resolve at L2 bandwidth.  After unrolling, (lane + 32j)>>6 == j>>1, so the
// per-head alpha index is compile-time constant.
// ---------------------------------------------------------------------------
__global__ void __launch_bounds__(256)
scatter_kernel(const int* __restrict__ ei, const float* __restrict__ ebuf,
               const float* __restrict__ ssum, const float* __restrict__ h,
               float* __restrict__ out) {
  int edge = (blockIdx.x << 3) + (threadIdx.x >> 5);
  if (edge >= N_EDGES) return;
  int lane = threadIdx.x & 31;
  int s = ei[edge], d = ei[N_EDGES + edge];
  const float* hs = h + (size_t)s * HF;
  float* od = out + (size_t)d * HF;
  __builtin_prefetch(hs + lane * 8, 0, 0);   // global_prefetch_b8 on h row
  v4f e4 = *(const v4f*)(ebuf + (size_t)edge * HEADS);
  v4f s4 = *(const v4f*)(ssum + (size_t)d * HEADS);
  v4f alpha;
  #pragma unroll
  for (int k = 0; k < 4; ++k) alpha[k] = e4[k] / (s4[k] + GAT_EPS);
  #pragma unroll
  for (int j = 0; j < 8; ++j) {
    int f = lane + (j << 5);
    float a = alpha[j >> 1];                 // head = f>>6 == j>>1 (const)
    atomicAdd(&od[f], a * hs[f]);
  }
}

// ---------------------------------------------------------------------------
extern "C" void kernel_launch(void* const* d_in, const int* in_sizes, int n_in,
                              void* d_out, int out_size, void* d_ws, size_t ws_size,
                              hipStream_t stream) {
  const float* x       = (const float*)d_in[0];
  const int*   ei      = (const int*)  d_in[1];
  const float* W       = (const float*)d_in[2];
  const float* att_src = (const float*)d_in[3];
  const float* att_dst = (const float*)d_in[4];
  const float* bias    = (const float*)d_in[5];
  float* out = (float*)d_out;

  // Carve workspace (~80 MB), 256 B aligned chunks
  char* p = (char*)d_ws;
  auto alloc = [&](size_t bytes) {
    char* r = p; p += (bytes + 255) & ~(size_t)255; return r;
  };
  float*    h     = (float*)   alloc((size_t)N_NODES * HF * 4);
  _Float16* Xh    = (_Float16*)alloc((size_t)N_NODES * IN_F * 2);
  _Float16* Wh    = (_Float16*)alloc((size_t)HF * IN_F * 2);
  float*    a_src = (float*)   alloc((size_t)N_NODES * HEADS * 4);
  float*    a_dst = (float*)   alloc((size_t)N_NODES * HEADS * 4);
  int*      m     = (int*)     alloc((size_t)N_NODES * HEADS * 4);
  float*    ssum  = (float*)   alloc((size_t)N_NODES * HEADS * 4);
  float*    ebuf  = (float*)   alloc((size_t)N_EDGES * HEADS * 4);

  cvt_f16_kernel  <<<2048, 256, 0, stream>>>(x, W, Xh, Wh);
  init_kernel     <<<2048, 256, 0, stream>>>(out, bias, m, ssum);
  gemm_wmma_kernel<<<6250, 256, 0, stream>>>(Xh, Wh, h);     // 50000 waves, 1 tile each
  logits_kernel   <<<6250, 256, 0, stream>>>(h, att_src, att_dst, a_src, a_dst);
  edge_score_kernel<<<(N_EDGES + 255) / 256, 256, 0, stream>>>(ei, a_src, a_dst, ebuf, m);
  edge_exp_kernel <<<(N_EDGES + 255) / 256, 256, 0, stream>>>(ei, m, ebuf, ssum);
  scatter_kernel  <<<N_EDGES / 8, 256, 0, stream>>>(ei, ebuf, ssum, h, out);
}
